// GINv2Model_61229053771856
// MI455X (gfx1250) — matrix operand order repaired and checked by reading, not compiled, not run on hardware
//
#include <hip/hip_runtime.h>
#include <hip/hip_bf16.h>

#define HDIM   128
#define LAYERS 5
#define GRAPHS 512
#define BN_EPS 1e-5f

typedef __attribute__((ext_vector_type(16))) __bf16 v16bf;
typedef __attribute__((ext_vector_type(8)))  float  v8f;

union Frag16 {
  v16bf v;
  unsigned short s[16];
  unsigned int   u[8];
};

__device__ __forceinline__ unsigned short f2bf(float f) {
  union { float f; unsigned int u; } c; c.f = f;
  // round-to-nearest-even fp32 -> bf16 (inputs are finite)
  return (unsigned short)((c.u + 0x7FFFu + ((c.u >> 16) & 1u)) >> 16);
}

__device__ __forceinline__ float bf2f(unsigned short b) {
  union { unsigned int u; float f; } c; c.u = ((unsigned int)b) << 16;
  return c.f;
}

enum { EPI_BIAS_RELU_BF16 = 0, EPI_BN_RELU = 1, EPI_EDGE = 2, EPI_ACCUM = 3 };
enum { AM_F32 = 0, AM_BF16 = 1, AM_EDGE = 2 };

// ---------------------------------------------------------------------------
// Generic fused GEMM: C[M,128] = epilogue(A[M,128] @ W[128,128] + bias)
//   A: fp32 / bf16 / synthesized edge embeddings; W: pre-converted bf16.
//   W staged transposed (N x K) in LDS so B-fragment loads are contiguous
//   dword LDS reads. Block: 256 threads = 8 waves; 64 rows x 128 cols of C.
//   wave = 2*rowTile + colHalf; each wave: 4 col-tiles x 4 K-chunks = 16 WMMAs
// ---------------------------------------------------------------------------
template <int EPI, int AMODE>
__global__ __launch_bounds__(256)
void gemm128_wmma(const float*          __restrict__ Af32,
                  const unsigned short* __restrict__ Abf,
                  const unsigned short* __restrict__ Wbf,   // [128,128] bf16 (k,n)
                  const float*          __restrict__ bias,  // [128]
                  float*                __restrict__ C,     // fp32 out (BN / ACCUM)
                  unsigned short*       __restrict__ Cbf,   // bf16 out (BIAS_RELU)
                  int M,
                  const float* __restrict__ bn_mean,
                  const float* __restrict__ bn_var,
                  const float* __restrict__ bn_gamma,
                  const float* __restrict__ bn_beta,
                  const int*   __restrict__ srcIdx,   // edge mode
                  const int*   __restrict__ dstIdx,
                  const float* __restrict__ xgather,  // x[N,128], gathered by src
                  float*       __restrict__ aggOut,   // agg[N,128], scatter-add
                  const int*   __restrict__ edge_cat, // [E,3]
                  const float* __restrict__ edge_emb) // [3,6,128]
{
  __shared__ unsigned short sA[64][132];   // rows x K, bf16 bits
  __shared__ unsigned short sBt[128][132]; // N x K (transposed), bf16 bits

  const int t = threadIdx.x;
  const int rowBase = blockIdx.x * 64;

  // ---- stage A tile (64 rows x 128 K) ------------------------------------
  {
    const int r  = t >> 2;          // 0..63
    const int cb = (t & 3) * 32;    // 0,32,64,96
    const int grow = rowBase + r;
    if (AMODE == AM_EDGE) {
      const float *e0 = edge_emb, *e1 = edge_emb, *e2 = edge_emb;
      const bool ok = (grow < M);
      if (ok) {
        int i0 = edge_cat[grow * 3 + 0]; i0 = i0 < 0 ? 0 : (i0 > 4 ? 4 : i0);
        int i1 = edge_cat[grow * 3 + 1]; i1 = i1 < 0 ? 0 : (i1 > 5 ? 5 : i1);
        int i2 = edge_cat[grow * 3 + 2]; i2 = i2 < 0 ? 0 : (i2 > 1 ? 1 : i2);
        e0 = edge_emb + (0 * 6 + i0) * HDIM;
        e1 = edge_emb + (1 * 6 + i1) * HDIM;
        e2 = edge_emb + (2 * 6 + i2) * HDIM;
      }
      #pragma unroll 8
      for (int c = 0; c < 32; ++c) {
        float v = ok ? (e0[cb + c] + e1[cb + c] + e2[cb + c]) : 0.0f;
        sA[r][cb + c] = f2bf(v);
      }
    } else if (AMODE == AM_BF16) {
      #pragma unroll
      for (int c = 0; c < 32; c += 8) {
        uint4 q = make_uint4(0u, 0u, 0u, 0u);
        if (grow < M) q = *(const uint4*)(Abf + (size_t)grow * HDIM + cb + c);
        const unsigned short* qs = (const unsigned short*)&q;
        #pragma unroll
        for (int j = 0; j < 8; ++j) sA[r][cb + c + j] = qs[j];
      }
    } else { // AM_F32
      #pragma unroll
      for (int c = 0; c < 32; c += 4) {
        float4 f = make_float4(0.f, 0.f, 0.f, 0.f);
        if (grow < M) f = *(const float4*)(Af32 + (size_t)grow * HDIM + cb + c);
        sA[r][cb + c + 0] = f2bf(f.x);
        sA[r][cb + c + 1] = f2bf(f.y);
        sA[r][cb + c + 2] = f2bf(f.z);
        sA[r][cb + c + 3] = f2bf(f.w);
      }
    }
  }
  // ---- stage W (128 K x 128 N) bf16, transposed --------------------------
  {
    const int k  = t >> 1;          // 0..127
    const int cb = (t & 1) * 64;    // 0 or 64
    #pragma unroll
    for (int c = 0; c < 64; c += 8) {
      uint4 q = *(const uint4*)(Wbf + (size_t)k * HDIM + cb + c);
      const unsigned short* qs = (const unsigned short*)&q;
      #pragma unroll
      for (int j = 0; j < 8; ++j) sBt[cb + c + j][k] = qs[j];
    }
  }

  // ---- prefetch random-gather targets so WMMA phase hides their latency --
  if (EPI == EPI_EDGE) {
    const int row = rowBase + (t >> 2);       // 64 rows, 4 threads/row
    if (row < M) {
      const float* pf = xgather + (size_t)srcIdx[row] * HDIM + (t & 3) * 32;
      __builtin_prefetch(pf, 0, 3);           // -> global_prefetch_b8
    }
  }
  __syncthreads();

  const int wave = t >> 5;
  const int lane = t & 31;
  const int half = lane >> 4;     // lane half
  const int l16  = lane & 15;
  const int rt   = wave >> 1;     // row tile 0..3
  const int p    = wave & 1;      // col half 0..1

  v8f acc[4];
  #pragma unroll
  for (int nt = 0; nt < 4; ++nt)
    #pragma unroll
    for (int r = 0; r < 8; ++r) acc[nt][r] = 0.0f;

  // Preload all 4 A fragments (ISA 7.12.2 16-bit A layout):
  //   e 0..7  -> K = 32kc + 8*half + e
  //   e 8..15 -> K = 32kc + 16 + 8*half + (e-8)
  const int arow = rt * 16 + l16;
  Frag16 afr[4];
  #pragma unroll
  for (int kc = 0; kc < 4; ++kc) {
    const unsigned int* ap0 =
        (const unsigned int*)&sA[arow][kc * 32 + 8 * half];
    const unsigned int* ap1 =
        (const unsigned int*)&sA[arow][kc * 32 + 16 + 8 * half];
    #pragma unroll
    for (int e = 0; e < 4; ++e) afr[kc].u[e]     = ap0[e];
    #pragma unroll
    for (int e = 0; e < 4; ++e) afr[kc].u[4 + e] = ap1[e];
  }

  #pragma unroll
  for (int kc = 0; kc < 4; ++kc) {
    #pragma unroll
    for (int nt = 0; nt < 4; ++nt) {
      // B 32x16 bf16 fragment: lane -> N = l16 (half selects K base),
      // e 0..7 -> K = 32kc + 8*half + e; e 8..15 -> K = 32kc + 16 + 8*half + (e-8)
      Frag16 b;
      const unsigned int* bp =
          (const unsigned int*)&sBt[(p * 4 + nt) * 16 + l16][kc * 32 + 8 * half];
      const unsigned int* bp1 =
          (const unsigned int*)&sBt[(p * 4 + nt) * 16 + l16][kc * 32 + 16 + 8 * half];
      #pragma unroll
      for (int e = 0; e < 4; ++e) { b.u[e] = bp[e]; b.u[4 + e] = bp1[e]; }

      acc[nt] = __builtin_amdgcn_wmma_f32_16x16x32_bf16(
          false, afr[kc].v, false, b.v, (short)0, acc[nt], false, false);
    }
  }

  // ---- epilogue: C layout M = r + 8*half, N = l16 (per 16x16 tile) -------
  #pragma unroll
  for (int nt = 0; nt < 4; ++nt) {
    const int col = (p * 4 + nt) * 16 + l16;
    const float bv = bias[col];
    float bm = 0.f, bs = 0.f, bg = 0.f, bb = 0.f;
    if (EPI == EPI_BN_RELU) {
      bm = bn_mean[col];
      bs = rsqrtf(bn_var[col] + BN_EPS);
      bg = bn_gamma[col];
      bb = bn_beta[col];
    }
    #pragma unroll
    for (int r = 0; r < 8; ++r) {
      const int row = rowBase + rt * 16 + half * 8 + r;
      if (row >= M) continue;
      float v = acc[nt][r] + bv;
      if (EPI == EPI_BIAS_RELU_BF16) {
        Cbf[(size_t)row * HDIM + col] = f2bf(fmaxf(v, 0.0f));
      } else if (EPI == EPI_BN_RELU) {
        C[(size_t)row * HDIM + col] = fmaxf((v - bm) * bs * bg + bb, 0.0f);
      } else if (EPI == EPI_EDGE) {
        const int s = srcIdx[row];
        const int d = dstIdx[row];
        v = fmaxf(v + xgather[(size_t)s * HDIM + col], 0.0f);
        atomicAdd(aggOut + (size_t)d * HDIM + col, v);
      } else { // EPI_ACCUM
        C[(size_t)row * HDIM + col] += v;
      }
    }
  }
}

// ---------------------------------------------------------------------------
// Elementwise / scatter helpers
// ---------------------------------------------------------------------------
__global__ void zero_kernel(float* __restrict__ p, long n) {
  long i = (long)blockIdx.x * blockDim.x + threadIdx.x;
  if (i < n) p[i] = 0.0f;
}

__global__ void f32_to_bf16_kernel(const float* __restrict__ s,
                                   unsigned short* __restrict__ d, long n) {
  long i = (long)blockIdx.x * blockDim.x + threadIdx.x;
  if (i < n) d[i] = f2bf(s[i]);
}

__global__ void encode_nodes_kernel(const int* __restrict__ x_cat,
                                    const float* __restrict__ emb, // [9,119,128]
                                    float* __restrict__ x, int N) {
  const int cards[9] = {119, 4, 12, 12, 10, 6, 6, 2, 2};
  long i = (long)blockIdx.x * blockDim.x + threadIdx.x;
  if (i >= (long)N * HDIM) return;
  int node = (int)(i >> 7), col = (int)(i & 127);
  float s = 0.0f;
  #pragma unroll
  for (int f = 0; f < 9; ++f) {
    int ix = x_cat[node * 9 + f];
    ix = ix < 0 ? 0 : (ix >= cards[f] ? cards[f] - 1 : ix);
    s += emb[((size_t)(f * 119 + ix)) * HDIM + col];
  }
  x[i] = s;
}

__global__ void count_kernel(const int* __restrict__ batch, float* __restrict__ cnt, int N) {
  int i = blockIdx.x * blockDim.x + threadIdx.x;
  if (i < N) atomicAdd(&cnt[batch[i]], 1.0f);
}

__global__ void inv_kernel(const float* __restrict__ cnt, float* __restrict__ invc, int G) {
  int i = blockIdx.x * blockDim.x + threadIdx.x;
  if (i < G) invc[i] = 1.0f / fmaxf(cnt[i], 1.0f);
}

__global__ void vn_init_kernel(const float* __restrict__ vn_emb, float* __restrict__ vn, int G) {
  long i = (long)blockIdx.x * blockDim.x + threadIdx.x;
  if (i < (long)G * HDIM) vn[i] = vn_emb[i & 127];
}

__global__ void add_vn_kernel(float* __restrict__ x, const float* __restrict__ vn,
                              const int* __restrict__ batch, int N) {
  long i = (long)blockIdx.x * blockDim.x + threadIdx.x;
  if (i >= (long)N * HDIM) return;
  x[i] += vn[(size_t)batch[i >> 7] * HDIM + (i & 127)];
}

// h(bf16) = (1+eps)*x + agg  -- consumed only as a WMMA A-operand, so bf16
// storage is lossless vs. converting during LDS staging.
__global__ void combine_kernel(unsigned short* __restrict__ h,
                               const float* __restrict__ x,
                               const float* __restrict__ agg,
                               const float* __restrict__ eps, int l, int N) {
  long i = (long)blockIdx.x * blockDim.x + threadIdx.x;
  if (i >= (long)N * HDIM) return;
  h[i] = f2bf((1.0f + eps[l]) * x[i] + agg[i]);
}

__global__ void pool_scatter_kernel(const float* __restrict__ x,
                                    const int* __restrict__ batch,
                                    float* __restrict__ g, int N) {
  long i = (long)blockIdx.x * blockDim.x + threadIdx.x;
  if (i >= (long)N * HDIM) return;
  atomicAdd(&g[(size_t)batch[i >> 7] * HDIM + (i & 127)], x[i]);
}

__global__ void scale_rows_kernel(float* __restrict__ g, const float* __restrict__ invc, int G) {
  long i = (long)blockIdx.x * blockDim.x + threadIdx.x;
  if (i < (long)G * HDIM) g[i] *= invc[i >> 7];
}

__global__ void head_final_kernel(const unsigned short* __restrict__ h1g, // bf16
                                  const float* __restrict__ w2,           // [128]
                                  const float* __restrict__ b2,           // [1]
                                  float* __restrict__ out, int G) {
  int gi = blockIdx.x * blockDim.x + threadIdx.x;
  if (gi >= G) return;
  float s = 0.0f;
  #pragma unroll 16
  for (int k = 0; k < HDIM; ++k) s += bf2f(h1g[(size_t)gi * HDIM + k]) * w2[k];
  out[gi] = s + b2[0];
}

// ---------------------------------------------------------------------------
extern "C" void kernel_launch(void* const* d_in, const int* in_sizes, int n_in,
                              void* d_out, int out_size, void* d_ws, size_t ws_size,
                              hipStream_t stream) {
  const int*   x_cat      = (const int*)d_in[0];
  const int*   edge_index = (const int*)d_in[1];
  const int*   edge_cat   = (const int*)d_in[2];
  const int*   batch      = (const int*)d_in[3];
  // d_in[4] = num_graphs (device scalar; G hardcoded = 512)
  const float* node_emb   = (const float*)d_in[5];
  const float* edge_emb   = (const float*)d_in[6];
  const float* eps        = (const float*)d_in[7];
  const float* conv_w1    = (const float*)d_in[8];
  const float* conv_b1    = (const float*)d_in[9];
  const float* conv_w2    = (const float*)d_in[10];
  const float* conv_b2    = (const float*)d_in[11];
  const float* edge_lin_w = (const float*)d_in[12];
  const float* edge_lin_b = (const float*)d_in[13];
  const float* bn_gamma   = (const float*)d_in[14];
  const float* bn_beta    = (const float*)d_in[15];
  const float* bn_mean    = (const float*)d_in[16];
  const float* bn_var     = (const float*)d_in[17];
  const float* vn_emb     = (const float*)d_in[18];
  const float* vn_w1      = (const float*)d_in[19];
  const float* vn_b1      = (const float*)d_in[20];
  const float* vn_w2      = (const float*)d_in[21];
  const float* vn_b2      = (const float*)d_in[22];
  const float* head_w1    = (const float*)d_in[23];
  const float* head_b1    = (const float*)d_in[24];
  const float* head_w2    = (const float*)d_in[25];
  const float* head_b2    = (const float*)d_in[26];

  const int N = in_sizes[0] / 9;
  const int E = in_sizes[1] / 2;
  const int G = GRAPHS;
  const int HH = HDIM * HDIM;
  const int* srcI = edge_index;
  const int* dstI = edge_index + E;

  // ---- workspace layout: fp32 region, then 256B-aligned bf16 region ------
  const size_t NH = (size_t)N * HDIM;
  const size_t GH = (size_t)G * HDIM;
  float* x    = (float*)d_ws;
  float* agg  = x   + NH;
  float* vn   = agg + NH;
  float* g    = vn  + GH;
  float* cnt  = g   + GH;
  float* invc = cnt + G;
  const size_t fcount = 2 * NH + 2 * GH + 2 * (size_t)G;
  const size_t bf_off = ((fcount * 4 + 255) / 256) * 256;
  unsigned short* h     = (unsigned short*)((char*)d_ws + bf_off);
  unsigned short* h1    = h    + NH;
  unsigned short* h1g   = h1   + NH;
  unsigned short* wedge = h1g  + GH;           // 5 x [128,128] bf16
  unsigned short* wc1   = wedge + 5 * HH;
  unsigned short* wc2   = wc1   + 5 * HH;
  unsigned short* wvn1  = wc2   + 5 * HH;
  unsigned short* wvn2  = wvn1  + HH;
  unsigned short* whd1  = wvn2  + HH;
  (void)ws_size; (void)n_in; (void)out_size;

  const unsigned TB = 256;
  auto gs = [](long n) { return (unsigned)((n + 255) / 256); };
  const unsigned gN = (unsigned)((N + 63) / 64);
  const unsigned gE = (unsigned)((E + 63) / 64);
  const unsigned gG = (unsigned)((G + 63) / 64);

  // ---- one-time per launch: weights -> bf16, encoders, graph constants ---
  f32_to_bf16_kernel<<<gs(5L * HH), TB, 0, stream>>>(edge_lin_w, wedge, 5L * HH);
  f32_to_bf16_kernel<<<gs(5L * HH), TB, 0, stream>>>(conv_w1, wc1, 5L * HH);
  f32_to_bf16_kernel<<<gs(5L * HH), TB, 0, stream>>>(conv_w2, wc2, 5L * HH);
  f32_to_bf16_kernel<<<gs(HH), TB, 0, stream>>>(vn_w1, wvn1, HH);
  f32_to_bf16_kernel<<<gs(HH), TB, 0, stream>>>(vn_w2, wvn2, HH);
  f32_to_bf16_kernel<<<gs(HH), TB, 0, stream>>>(head_w1, whd1, HH);

  encode_nodes_kernel<<<gs(NH), TB, 0, stream>>>(x_cat, node_emb, x, N);
  zero_kernel<<<gs(G), TB, 0, stream>>>(cnt, G);
  count_kernel<<<gs(N), TB, 0, stream>>>(batch, cnt, N);
  inv_kernel<<<gs(G), TB, 0, stream>>>(cnt, invc, G);
  vn_init_kernel<<<gs(GH), TB, 0, stream>>>(vn_emb, vn, G);

  for (int l = 0; l < LAYERS; ++l) {
    const unsigned short* ew = wedge + (size_t)l * HH;
    const unsigned short* w1 = wc1   + (size_t)l * HH;
    const unsigned short* w2 = wc2   + (size_t)l * HH;
    const float* eb = edge_lin_b + (size_t)l * HDIM;
    const float* b1 = conv_b1 + (size_t)l * HDIM;
    const float* b2 = conv_b2 + (size_t)l * HDIM;
    const float* bm = bn_mean + (size_t)l * HDIM;
    const float* bv = bn_var + (size_t)l * HDIM;
    const float* bg = bn_gamma + (size_t)l * HDIM;
    const float* bb = bn_beta + (size_t)l * HDIM;

    // x += vn[batch]
    add_vn_kernel<<<gs(NH), TB, 0, stream>>>(x, vn, batch, N);

    // agg = scatter_add(relu(x[src] + e @ ew + eb), dst)  (fully fused)
    zero_kernel<<<gs(NH), TB, 0, stream>>>(agg, (long)NH);
    gemm128_wmma<EPI_EDGE, AM_EDGE><<<gE, TB, 0, stream>>>(
        nullptr, nullptr, ew, eb, nullptr, nullptr, E,
        nullptr, nullptr, nullptr, nullptr,
        srcI, dstI, x, agg, edge_cat, edge_emb);

    // h = (1+eps)*x + agg;  h1 = relu(h@w1+b1);  x = relu(BN(h1@w2+b2))
    combine_kernel<<<gs(NH), TB, 0, stream>>>(h, x, agg, eps, l, N);
    gemm128_wmma<EPI_BIAS_RELU_BF16, AM_BF16><<<gN, TB, 0, stream>>>(
        nullptr, h, w1, b1, nullptr, h1, N,
        nullptr, nullptr, nullptr, nullptr,
        nullptr, nullptr, nullptr, nullptr, nullptr, nullptr);
    gemm128_wmma<EPI_BN_RELU, AM_BF16><<<gN, TB, 0, stream>>>(
        nullptr, h1, w2, b2, x, nullptr, N,
        bm, bv, bg, bb,
        nullptr, nullptr, nullptr, nullptr, nullptr, nullptr);

    if (l < LAYERS - 1) {
      // g = mean_pool(x);  vn += relu(g@vn_w1+vn_b1) @ vn_w2 + vn_b2
      zero_kernel<<<gs(GH), TB, 0, stream>>>(g, (long)GH);
      pool_scatter_kernel<<<gs(NH), TB, 0, stream>>>(x, batch, g, N);
      scale_rows_kernel<<<gs(GH), TB, 0, stream>>>(g, invc, G);
      gemm128_wmma<EPI_BIAS_RELU_BF16, AM_F32><<<gG, TB, 0, stream>>>(
          g, nullptr, wvn1, vn_b1, nullptr, h1g, G,
          nullptr, nullptr, nullptr, nullptr,
          nullptr, nullptr, nullptr, nullptr, nullptr, nullptr);
      gemm128_wmma<EPI_ACCUM, AM_BF16><<<gG, TB, 0, stream>>>(
          nullptr, h1g, wvn2, vn_b2, vn, nullptr, G,
          nullptr, nullptr, nullptr, nullptr,
          nullptr, nullptr, nullptr, nullptr, nullptr, nullptr);
    }
  }

  // head: g = mean_pool(x); logits = relu(g@head_w1+b1) @ head_w2 + b2
  zero_kernel<<<gs(GH), TB, 0, stream>>>(g, (long)GH);
  pool_scatter_kernel<<<gs(NH), TB, 0, stream>>>(x, batch, g, N);
  scale_rows_kernel<<<gs(GH), TB, 0, stream>>>(g, invc, G);
  gemm128_wmma<EPI_BIAS_RELU_BF16, AM_F32><<<gG, TB, 0, stream>>>(
      g, nullptr, whd1, head_b1, nullptr, h1g, G,
      nullptr, nullptr, nullptr, nullptr,
      nullptr, nullptr, nullptr, nullptr, nullptr, nullptr);
  head_final_kernel<<<gs(G), TB, 0, stream>>>(h1g, head_w2, head_b2, (float*)d_out, G);
}
